// MyAttention3_37185826849034
// MI455X (gfx1250) — matrix operand order repaired and checked by reading, not compiled, not run on hardware
//
#include <hip/hip_runtime.h>

// ---------------------------------------------------------------------------
// Gathered top-k attention for MI455X (gfx1250, wave32, WMMA).
//   query [B,Lq,H,C] f32, key/value [B,Lk,H,C] f32, pos [B,Lq,H,K] i32
//   out   [B,Lq,H,C] f32.  B=2, Lq=Lk=2048, H=16, C=64, K=64.
// One wave per (b,l,h). QK and PV both run on v_wmma_f32_16x16x32_bf16.
// ---------------------------------------------------------------------------

typedef __attribute__((ext_vector_type(16))) __bf16 v16bf;
typedef __attribute__((ext_vector_type(8)))  __bf16 v8bf;
typedef __attribute__((ext_vector_type(8)))  float  v8f;

#define WAVES_PER_BLOCK 4

constexpr int kLq   = 2048;
constexpr int kLk   = 2048;
constexpr int kH    = 16;
constexpr int kC    = 64;   // head dim
constexpr int kK    = 64;   // top-k per query
constexpr int kRowStride = kH * kC;   // stride (floats) between kv rows j and j+1
constexpr int kVT  = 72;              // padded LDS stride (halves) per channel row

__device__ __forceinline__ v8f wmma_bf16(v16bf a, v16bf b, v8f c) {
  // (neg_a, A, neg_b, B, c_mod, C, reuse_a, reuse_b)
  return __builtin_amdgcn_wmma_f32_16x16x32_bf16(false, a, false, b, (short)0, c,
                                                 false, false);
}

// Pack 16 fp32 (two contiguous 8-float segments p0, p0+16) into a bf16 A-frag.
// A-matrix 16-bit layout (ISA 7.12.2): lane half 0 -> K {0..7,16..23},
// lane half 1 -> K {8..15,24..31} of the 32-wide chunk; base points at the
// lane's first segment already.
__device__ __forceinline__ v16bf pack_a_global(const float* base) {
  const float4 a0 = ((const float4*)base)[0];
  const float4 a1 = ((const float4*)base)[1];
  const float4 b0 = ((const float4*)(base + 16))[0];
  const float4 b1 = ((const float4*)(base + 16))[1];
  v16bf f;
  f[0] = (__bf16)a0.x; f[1] = (__bf16)a0.y; f[2]  = (__bf16)a0.z; f[3]  = (__bf16)a0.w;
  f[4] = (__bf16)a1.x; f[5] = (__bf16)a1.y; f[6]  = (__bf16)a1.z; f[7]  = (__bf16)a1.w;
  f[8] = (__bf16)b0.x; f[9] = (__bf16)b0.y; f[10] = (__bf16)b0.z; f[11] = (__bf16)b0.w;
  f[12] = (__bf16)b1.x; f[13] = (__bf16)b1.y; f[14] = (__bf16)b1.z; f[15] = (__bf16)b1.w;
  return f;
}

__global__ __launch_bounds__(32 * WAVES_PER_BLOCK)
void topk_attn_wmma(const float* __restrict__ query,
                    const float* __restrict__ key,
                    const float* __restrict__ value,
                    const int*   __restrict__ pos,
                    float*       __restrict__ out,
                    int num_qh) {
  __shared__ int    sPos[WAVES_PER_BLOCK][kK];
  __shared__ __bf16 sVT [WAVES_PER_BLOCK][kC * kVT];   // V^T, bf16, padded rows

  if ((int)(blockIdx.x * WAVES_PER_BLOCK) >= num_qh) return;  // block-uniform

  const int lane = threadIdx.x & 31;
  const int warp = threadIdx.x >> 5;
  const int hf   = lane >> 4;        // lane half (0: lanes 0-15, 1: lanes 16-31)
  const int lm   = lane & 15;
  int qh = blockIdx.x * WAVES_PER_BLOCK + warp;
  if (qh >= num_qh) qh = num_qh - 1;                  // redundant work, no divergence

  const int b = qh / (kLq * kH);
  const int h = qh % kH;
  const float* qrow  = query + (size_t)qh * kC;
  const int*   prow  = pos   + (size_t)qh * kK;
  const float* kbase = key   + (size_t)b * kLk * kRowStride + (size_t)h * kC;
  const float* vbase = value + (size_t)b * kLk * kRowStride + (size_t)h * kC;

  // ---- stage pos in LDS (64 ints per wave) --------------------------------
  sPos[warp][2 * lane]     = prow[2 * lane];
  sPos[warp][2 * lane + 1] = prow[2 * lane + 1];
  __syncthreads();

  // ---- build q B-fragments (broadcast across all 16 N columns) ------------
  // B-matrix 16-bit layout (matches A table): lane half 0 holds
  // K = chunk*32 + {0..7,16..23}, half 1 holds K = chunk*32 + {8..15,24..31}.
  v16bf qf[2];
#pragma unroll
  for (int ch = 0; ch < 2; ++ch) {
    const float* qp = qrow + ch * 32 + hf * 8;
    const float4 q0 = ((const float4*)qp)[0];
    const float4 q1 = ((const float4*)qp)[1];
    const float4 q2 = ((const float4*)(qp + 16))[0];
    const float4 q3 = ((const float4*)(qp + 16))[1];
    v16bf f;
    f[0]  = (__bf16)q0.x; f[1]  = (__bf16)q0.y; f[2]  = (__bf16)q0.z; f[3]  = (__bf16)q0.w;
    f[4]  = (__bf16)q1.x; f[5]  = (__bf16)q1.y; f[6]  = (__bf16)q1.z; f[7]  = (__bf16)q1.w;
    f[8]  = (__bf16)q2.x; f[9]  = (__bf16)q2.y; f[10] = (__bf16)q2.z; f[11] = (__bf16)q2.w;
    f[12] = (__bf16)q3.x; f[13] = (__bf16)q3.y; f[14] = (__bf16)q3.z; f[15] = (__bf16)q3.w;
    qf[ch] = f;
  }

  // ---- QK: scores for 4 groups of 16 gathered keys via WMMA ---------------
  // Lanes m and m+16 both own key row pos[g*16+m]; each loads its half's
  // column segments once -> each 256B key row is fetched exactly once/wave.
  v8f acc[4];
#pragma unroll
  for (int g = 0; g < 4; ++g) {
    v8f a = {0.f, 0.f, 0.f, 0.f, 0.f, 0.f, 0.f, 0.f};
    const int row = sPos[warp][g * 16 + lm];
    const float* kr = kbase + (size_t)row * kRowStride;
#pragma unroll
    for (int ch = 0; ch < 2; ++ch) {
      const v16bf af = pack_a_global(kr + ch * 32 + hf * 8);
      a = wmma_bf16(af, qf[ch], a);
    }
    acc[g] = a;
  }

  // ---- softmax over 64 scores --------------------------------------------
  // With broadcast B, lane half 0 holds rows 0-7 of every group in its D
  // VGPRs, half 1 holds rows 8-15; one xor-16 shuffle merges the halves.
  const float scale = 0.125f;  // 1/sqrt(64)
  float mx = -1e30f;
#pragma unroll
  for (int g = 0; g < 4; ++g)
#pragma unroll
    for (int r = 0; r < 8; ++r) {
      acc[g][r] *= scale;
      mx = fmaxf(mx, acc[g][r]);
    }
  mx = fmaxf(mx, __shfl_xor(mx, 16));

  float p[4][8];
  float sum = 0.f;
#pragma unroll
  for (int g = 0; g < 4; ++g)
#pragma unroll
    for (int r = 0; r < 8; ++r) {
      p[g][r] = __expf(acc[g][r] - mx);
      sum += p[g][r];
    }
  sum += __shfl_xor(sum, 16);
  const float inv = 1.f / sum;
#pragma unroll
  for (int g = 0; g < 4; ++g)
#pragma unroll
    for (int r = 0; r < 8; ++r) p[g][r] *= inv;

  // ---- gather V, store transposed bf16 into LDS ---------------------------
  // 32 iters: 2 rows/iter, 16 lanes per row, one float4 (4 channels) each.
#pragma unroll 4
  for (int i = 0; i < 32; ++i) {
    const int j   = i * 2 + hf;
    const int row = sPos[warp][j];
    const float4 v4 = *(const float4*)(vbase + (size_t)row * kRowStride + lm * 4);
    __bf16* d = &sVT[warp][0];
    d[(lm * 4 + 0) * kVT + j] = (__bf16)v4.x;
    d[(lm * 4 + 1) * kVT + j] = (__bf16)v4.y;
    d[(lm * 4 + 2) * kVT + j] = (__bf16)v4.z;
    d[(lm * 4 + 3) * kVT + j] = (__bf16)v4.w;
  }
  __syncthreads();

  // ---- build w B-fragments (broadcast weights) ----------------------------
  // Need w[j] for j = ch*32 + hf*16 + t (t=0..15). Rows 0-7 live in half 0,
  // rows 8-15 in half 1; exchange the missing 8 via xor-16 shuffles.
  v16bf wf[2];
#pragma unroll
  for (int ch = 0; ch < 2; ++ch) {
    v16bf f;
#pragma unroll
    for (int t = 0; t < 8; ++t) {
      const float o0 = __shfl_xor(p[2 * ch][t], 16);      // partner, group 2ch
      const float o1 = __shfl_xor(p[2 * ch + 1][t], 16);  // partner, group 2ch+1
      const float lo = hf ? o1 : p[2 * ch][t];            // rows 0-7 of own group
      const float hi = hf ? p[2 * ch + 1][t] : o0;        // rows 8-15 of own group
      f[t]     = (__bf16)lo;
      f[8 + t] = (__bf16)hi;
    }
    wf[ch] = f;
  }

  // ---- PV: out = V^T(64x64) * w via WMMA ----------------------------------
  v8f oacc[4];
#pragma unroll
  for (int gc = 0; gc < 4; ++gc) {
    v8f a = {0.f, 0.f, 0.f, 0.f, 0.f, 0.f, 0.f, 0.f};
    const int c = gc * 16 + lm;                 // channel (A-matrix row)
#pragma unroll
    for (int ch = 0; ch < 2; ++ch) {
      const __bf16* src = &sVT[warp][c * kVT + ch * 32 + hf * 16];
      const v8bf x0 = *(const v8bf*)src;         // ds_load_b128, 16B aligned
      const v8bf x1 = *(const v8bf*)(src + 8);
      v16bf af;
#pragma unroll
      for (int t = 0; t < 8; ++t) { af[t] = x0[t]; af[8 + t] = x1[t]; }
      a = wmma_bf16(af, wf[ch], a);
    }
    oacc[gc] = a;
  }

  // ---- store: D columns are duplicated; lanes 0 and 16 write 32 floats ----
  float* orow = out + (size_t)qh * kC;
  if (lm == 0) {
#pragma unroll
    for (int gc = 0; gc < 4; ++gc) {
      float* op = orow + gc * 16 + hf * 8;     // hf=0: rows 0-7, hf=1: rows 8-15
      ((float4*)op)[0] = make_float4(oacc[gc][0], oacc[gc][1], oacc[gc][2], oacc[gc][3]);
      ((float4*)op)[1] = make_float4(oacc[gc][4], oacc[gc][5], oacc[gc][6], oacc[gc][7]);
    }
  }
}

extern "C" void kernel_launch(void* const* d_in, const int* in_sizes, int n_in,
                              void* d_out, int out_size, void* d_ws, size_t ws_size,
                              hipStream_t stream) {
  const float* query = (const float*)d_in[0];
  const float* key   = (const float*)d_in[1];
  const float* value = (const float*)d_in[2];
  const int*   pos   = (const int*)d_in[3];
  float*       out   = (float*)d_out;

  const int num_qh = in_sizes[0] / kC;  // B*Lq*H = 65536
  const int blocks = (num_qh + WAVES_PER_BLOCK - 1) / WAVES_PER_BLOCK;
  topk_attn_wmma<<<blocks, 32 * WAVES_PER_BLOCK, 0, stream>>>(
      query, key, value, pos, out, num_qh);
}
